// DGRec_56556129353739
// MI455X (gfx1250) — matrix operand order repaired and checked by reading, not compiled, not run on hardware
//
#include <hip/hip_runtime.h>
#include <hip/hip_bf16.h>
#include <math.h>

typedef __attribute__((ext_vector_type(16))) _Float16 v16h;
typedef __attribute__((ext_vector_type(8)))  _Float16 v8h;
typedef __attribute__((ext_vector_type(8)))  float    v8f;

#define DV 128

// ---------------- WMMA fragment helpers (16x16x32 f16 -> f32) ----------------
// A: 16x32 (MxK). Lane l: m=l&15, half=l>>4. elems e<8 -> k=kbase+8*half+e,
//    e>=8 -> k=kbase+16+8*half+(e-8). Two contiguous 16B loads.
__device__ __forceinline__ v16h frag_a(const _Float16* rowbase, int kbase, int half) {
  const v8h* p0 = (const v8h*)(rowbase + kbase + 8 * half);
  const v8h* p1 = (const v8h*)(rowbase + kbase + 16 + 8 * half);
  v8h lo = *p0, hi = *p1;
  v16h r;
#pragma unroll
  for (int e = 0; e < 8; ++e) { r[e] = lo[e]; r[8 + e] = hi[e]; }
  return r;
}
// B: 32x16 (KxN). Lane l: n=l&15, half=l>>4; elems e<8 -> k=kbase+8*half+e,
// e>=8 -> k=kbase+16+8*half+(e-8). colbase = pointer to weight row j0+n (B column). Two contiguous 16B loads.
__device__ __forceinline__ v16h frag_b(const _Float16* colbase, int kbase, int half) {
  const v8h* p0 = (const v8h*)(colbase + kbase + 8 * half);
  const v8h* p1 = (const v8h*)(colbase + kbase + 16 + 8 * half);
  v8h lo = *p0, hi = *p1;
  v16h r;
#pragma unroll
  for (int e = 0; e < 8; ++e) { r[e] = lo[e]; r[8 + e] = hi[e]; }
  return r;
}
__device__ __forceinline__ v8f wmma_f16(v16h a, v16h b, v8f c) {
  return __builtin_amdgcn_wmma_f32_16x16x32_f16(false, a, false, b, (short)0, c, false, false);
}

// ---------------- small prep kernels ----------------
__global__ void fill_f32(float* p, float v, int n) {
  int i = blockIdx.x * 256 + threadIdx.x;
  if (i < n) p[i] = v;
}
__global__ void cvt_f32_to_f16(const float* __restrict__ s, _Float16* __restrict__ d, int n) {
  int i = blockIdx.x * 256 + threadIdx.x;
  if (i < n) d[i] = (_Float16)s[i];
}
__global__ void build_wcat(const float* __restrict__ Wih, const float* __restrict__ Whh,
                           _Float16* __restrict__ Wcat) {
  int i = blockIdx.x * 256 + threadIdx.x;
  if (i >= 512 * 256) return;
  int j = i >> 8, k = i & 255;
  float v = (k < 128) ? Wih[j * 128 + k] : Whh[j * 128 + (k - 128)];
  Wcat[i] = (_Float16)v;
}
__global__ void bias_sum(const float* a, const float* b, float* o, int n) {
  int i = blockIdx.x * 256 + threadIdx.x;
  if (i < n) o[i] = a[i] + b[i];
}
// renorm rows of a [nrows x 128] f32 matrix into f16
__global__ void renorm_rows_f16(const float* __restrict__ src, _Float16* __restrict__ dst, int nrows) {
  int m = threadIdx.x >> 4, lg = threadIdx.x & 15;
  int row = blockIdx.x * 16 + m;
  if (row >= nrows) return;
  const float* r = src + (size_t)row * DV;
  int dbase = lg * 8;
  float v[8], ss = 0.f;
#pragma unroll
  for (int j = 0; j < 8; ++j) { v[j] = r[dbase + j]; ss += v[j] * v[j]; }
#pragma unroll
  for (int o = 8; o > 0; o >>= 1) ss += __shfl_xor(ss, o, 16);
  float sc = fminf(1.f, 1.f / fmaxf(sqrtf(ss), 1e-12f));
#pragma unroll
  for (int j = 0; j < 8; ++j) dst[(size_t)row * DV + dbase + j] = (_Float16)(v[j] * sc);
}

// ---------------- fused LSTM (gather + renorm + WMMA GEMM + cell) ----------------
__global__ __launch_bounds__(256) void lstm_kernel(
    const float* __restrict__ item_emb, const int* __restrict__ seqs,
    const int* __restrict__ lens, const _Float16* __restrict__ Wcat,
    const float* __restrict__ bias, float* __restrict__ hn_out, int T) {
  __shared__ _Float16 sh_x[16][256];   // [x_t(renorm) | h] in f16
  __shared__ float sh_g[16][512];      // gates f32
  const int tid = threadIdx.x;
  const int r0 = blockIdx.x * 16;
  const int m = tid >> 4, lg = tid & 15;        // elementwise mapping
  const int w = tid >> 5, lane = tid & 31;      // wave mapping
  const int half = lane >> 4, nlo = lane & 15;
  const int dbase = lg * 8;
  float c_reg[8], h_reg[8];
#pragma unroll
  for (int j = 0; j < 8; ++j) { c_reg[j] = 0.f; h_reg[j] = 0.f; }
  const int len = lens[r0 + m];
#pragma unroll
  for (int j = 0; j < 8; ++j) sh_x[m][128 + dbase + j] = (_Float16)0.f;

  for (int t = 0; t < T; ++t) {
    // gather x_t = renorm(item_emb[seq]) with item 0 forced to zero
    int item = seqs[(size_t)(r0 + m) * T + t];
    const float* erow = item_emb + (size_t)item * DV;
    float xv[8], ss = 0.f;
#pragma unroll
    for (int j = 0; j < 8; ++j) {
      float v = (item == 0) ? 0.f : erow[dbase + j];
      xv[j] = v; ss += v * v;
    }
#pragma unroll
    for (int o = 8; o > 0; o >>= 1) ss += __shfl_xor(ss, o, 16);
    float sc = fminf(1.f, 1.f / fmaxf(sqrtf(ss), 1e-12f));
#pragma unroll
    for (int j = 0; j < 8; ++j) sh_x[m][dbase + j] = (_Float16)(xv[j] * sc);
    __syncthreads();

    // gates[16x512] = [x|h][16x256] @ Wcat^T, 8 waves x 4 tiles
    const _Float16* arow = &sh_x[nlo][0];
#pragma unroll
    for (int q = 0; q < 4; ++q) {
      const int j0 = (w * 4 + q) * 16;
      const _Float16* bcol = Wcat + (size_t)(j0 + nlo) * 256;
      v8f acc = {};
#pragma unroll
      for (int kb = 0; kb < 8; ++kb) {
        v16h af = frag_a(arow, kb * 32, half);
        v16h bf = frag_b(bcol, kb * 32, half);
        acc = wmma_f16(af, bf, acc);
      }
#pragma unroll
      for (int v = 0; v < 8; ++v) sh_g[v + 8 * half][j0 + nlo] = acc[v];
    }
    __syncthreads();

    // LSTM cell, masked by t < len; h kept in regs + mirrored to sh_x f16
#pragma unroll
    for (int j = 0; j < 8; ++j) {
      int d = dbase + j;
      float gi = sh_g[m][d]       + bias[d];
      float gf = sh_g[m][128 + d] + bias[128 + d];
      float gg = sh_g[m][256 + d] + bias[256 + d];
      float go = sh_g[m][384 + d] + bias[384 + d];
      float si = 1.f / (1.f + expf(-gi));
      float sf = 1.f / (1.f + expf(-gf));
      float so = 1.f / (1.f + expf(-go));
      float cn = sf * c_reg[j] + si * tanhf(gg);
      float hv = so * tanhf(cn);
      if (t < len) { c_reg[j] = cn; h_reg[j] = hv; }
      sh_x[m][128 + d] = (_Float16)h_reg[j];
    }
    __syncthreads();
  }
#pragma unroll
  for (int j = 0; j < 8; ++j) hn_out[(size_t)(r0 + m) * DV + dbase + j] = h_reg[j];
}

// ---------------- feat = relu([renorm(user), hn] @ W1^T) ----------------
__global__ __launch_bounds__(256) void feat_kernel(
    const float* __restrict__ user_emb, const int* __restrict__ uids,
    const float* __restrict__ hn, const _Float16* __restrict__ W1h,
    float* __restrict__ feat) {
  __shared__ _Float16 a[16][256];
  const int tid = threadIdx.x, r0 = blockIdx.x * 16;
  const int m = tid >> 4, lg = tid & 15;
  const int w = tid >> 5, lane = tid & 31;
  const int half = lane >> 4, nlo = lane & 15;
  const int dbase = lg * 8;
  int uid = uids[r0 + m];
  const float* ur = user_emb + (size_t)uid * DV;
  float v[8], ss = 0.f;
#pragma unroll
  for (int j = 0; j < 8; ++j) { v[j] = ur[dbase + j]; ss += v[j] * v[j]; }
#pragma unroll
  for (int o = 8; o > 0; o >>= 1) ss += __shfl_xor(ss, o, 16);
  float sc = fminf(1.f, 1.f / fmaxf(sqrtf(ss), 1e-12f));
#pragma unroll
  for (int j = 0; j < 8; ++j) {
    a[m][dbase + j] = (_Float16)(v[j] * sc);
    a[m][128 + dbase + j] = (_Float16)hn[(size_t)(r0 + m) * DV + dbase + j];
  }
  __syncthreads();
  const int j0 = w * 16;
  const _Float16* arow = &a[nlo][0];
  const _Float16* bcol = W1h + (size_t)(j0 + nlo) * 256;
  v8f acc = {};
#pragma unroll
  for (int kb = 0; kb < 8; ++kb)
    acc = wmma_f16(frag_a(arow, kb * 32, half), frag_b(bcol, kb * 32, half), acc);
#pragma unroll
  for (int vv = 0; vv < 8; ++vv)
    feat[(size_t)(r0 + vv + 8 * half) * DV + j0 + nlo] = fmaxf(acc[vv], 0.f);
}

// cur = hn[cur_sidx]; feat[cur_sidx] = cur
__global__ void cur_kernel(const float* __restrict__ hn, const int* __restrict__ sidx,
                           float* __restrict__ cur, float* __restrict__ feat, int N2) {
  int i = blockIdx.x * 256 + threadIdx.x;
  if (i >= N2 * DV) return;
  int r = i >> 7, d = i & 127;
  int s = sidx[r];
  float v = hn[(size_t)s * DV + d];
  cur[i] = v;
  feat[(size_t)s * DV + d] = v;
}

// ---------------- GAT segment ops ----------------
__device__ __forceinline__ void atomicMaxF32(float* a, float v) {
  if (v >= 0.f) atomicMax((int*)a, __float_as_int(v));
  else atomicMin((unsigned int*)a, __float_as_uint(v));
}
__global__ void edge_score_kernel(const float* __restrict__ feat, const int* __restrict__ src,
                                  const int* __restrict__ dst, const int* __restrict__ idx_map,
                                  float* __restrict__ score, float* __restrict__ mbuf, int E) {
  int e = (blockIdx.x * 256 + threadIdx.x) >> 5;
  int lane = threadIdx.x & 31;
  if (e >= E) return;
  int s = src[e], d = dst[e];
  const float* fs = feat + (size_t)s * DV;
  const float* fd = feat + (size_t)idx_map[d] * DV;
  float acc = 0.f;
#pragma unroll
  for (int j = 0; j < 4; ++j) { int k = lane * 4 + j; acc += fs[k] * fd[k]; }
#pragma unroll
  for (int o = 16; o > 0; o >>= 1) acc += __shfl_xor(acc, o, 32);
  if (lane == 0) { score[e] = acc; atomicMaxF32(&mbuf[d], acc); }
}
__global__ void edge_w_kernel(const float* __restrict__ score, const int* __restrict__ dst,
                              const float* __restrict__ mbuf, float* __restrict__ wbuf,
                              float* __restrict__ zbuf, int E) {
  int e = blockIdx.x * 256 + threadIdx.x;
  if (e >= E) return;
  int d = dst[e];
  float wv = expf(score[e] - mbuf[d]);
  wbuf[e] = wv;
  atomicAdd(&zbuf[d], wv);
}
__global__ void edge_agg_kernel(const float* __restrict__ feat, const int* __restrict__ src,
                                const int* __restrict__ dst, const float* __restrict__ wbuf,
                                const float* __restrict__ zbuf, float* __restrict__ agg, int E) {
  int e = (blockIdx.x * 256 + threadIdx.x) >> 5;
  int lane = threadIdx.x & 31;
  if (e >= E) return;
  int s = src[e], d = dst[e];
  float wn = wbuf[e] / fmaxf(zbuf[d], 1e-12f);
  const float* fs = feat + (size_t)s * DV;
#pragma unroll
  for (int j = 0; j < 4; ++j) {
    int k = lane * 4 + j;
    atomicAdd(&agg[(size_t)d * DV + k], wn * fs[k]);
  }
}
// out = feat_in[idx_map] + relu(agg @ W^T + b)   (WMMA, K=128)
__global__ __launch_bounds__(256) void gat_out_kernel(
    const float* __restrict__ agg, const float* __restrict__ feat_in,
    const int* __restrict__ idx_map, const _Float16* __restrict__ Wh,
    const float* __restrict__ b, float* __restrict__ out) {
  __shared__ _Float16 a[16][128];
  const int tid = threadIdx.x, r0 = blockIdx.x * 16;
  const int m = tid >> 4, lg = tid & 15;
  const int w = tid >> 5, lane = tid & 31;
  const int half = lane >> 4, nlo = lane & 15;
  const int dbase = lg * 8;
#pragma unroll
  for (int j = 0; j < 8; ++j)
    a[m][dbase + j] = (_Float16)agg[(size_t)(r0 + m) * DV + dbase + j];
  __syncthreads();
  const int j0 = w * 16;
  const _Float16* arow = &a[nlo][0];
  const _Float16* bcol = Wh + (size_t)(j0 + nlo) * DV;
  v8f acc = {};
#pragma unroll
  for (int kb = 0; kb < 4; ++kb)
    acc = wmma_f16(frag_a(arow, kb * 32, half), frag_b(bcol, kb * 32, half), acc);
#pragma unroll
  for (int vv = 0; vv < 8; ++vv) {
    int row = r0 + vv + 8 * half;
    float rst = fmaxf(acc[vv] + b[j0 + nlo], 0.f);
    out[(size_t)row * DV + j0 + nlo] = feat_in[(size_t)idx_map[row] * DV + j0 + nlo] + rst;
  }
}

// sr(f16) = [cur, feat2] @ W2^T
__global__ __launch_bounds__(256) void sr_kernel(
    const float* __restrict__ cur, const float* __restrict__ feat2,
    const _Float16* __restrict__ W2h, _Float16* __restrict__ srh) {
  __shared__ _Float16 a[16][256];
  const int tid = threadIdx.x, r0 = blockIdx.x * 16;
  const int m = tid >> 4, lg = tid & 15;
  const int w = tid >> 5, lane = tid & 31;
  const int half = lane >> 4, nlo = lane & 15;
  const int dbase = lg * 8;
#pragma unroll
  for (int j = 0; j < 8; ++j) {
    a[m][dbase + j]       = (_Float16)cur[(size_t)(r0 + m) * DV + dbase + j];
    a[m][128 + dbase + j] = (_Float16)feat2[(size_t)(r0 + m) * DV + dbase + j];
  }
  __syncthreads();
  const int j0 = w * 16;
  const _Float16* arow = &a[nlo][0];
  const _Float16* bcol = W2h + (size_t)(j0 + nlo) * 256;
  v8f acc = {};
#pragma unroll
  for (int kb = 0; kb < 8; ++kb)
    acc = wmma_f16(frag_a(arow, kb * 32, half), frag_b(bcol, kb * 32, half), acc);
#pragma unroll
  for (int vv = 0; vv < 8; ++vv)
    srh[(size_t)(r0 + vv + 8 * half) * DV + j0 + nlo] = (_Float16)acc[vv];
}

// logits[M x N] = sr @ it1n^T, K=128, straight-from-global fragments
__global__ __launch_bounds__(256) void logits_kernel(
    const _Float16* __restrict__ sr, const _Float16* __restrict__ it1n,
    float* __restrict__ out, int N) {
  const int tid = threadIdx.x, r0 = blockIdx.x * 16;
  const int w = tid >> 5, lane = tid & 31;
  const int half = lane >> 4, nlo = lane & 15;
  const int c = blockIdx.y * 128 + w * 16 + nlo;
  const bool cv = c < N;
  const _Float16* arow = sr + (size_t)(r0 + nlo) * DV;
  const _Float16* bcol = it1n + (size_t)(cv ? c : 0) * DV;
  v8f acc = {};
#pragma unroll
  for (int kb = 0; kb < 4; ++kb) {
    v16h af = frag_a(arow, kb * 32, half);
    v16h bf = frag_b(bcol, kb * 32, half);
    if (!cv) { v16h z = {}; bf = z; }
    acc = wmma_f16(af, bf, acc);
  }
  if (cv) {
#pragma unroll
    for (int vv = 0; vv < 8; ++vv)
      out[(size_t)(r0 + vv + 8 * half) * N + c] = acc[vv];
  }
}

extern "C" void kernel_launch(void* const* d_in, const int* in_sizes, int n_in,
                              void* d_out, int out_size, void* d_ws, size_t ws_size,
                              hipStream_t stream) {
  const float* user_emb = (const float*)d_in[0];
  const float* item_emb = (const float*)d_in[1];
  const float* Wih = (const float*)d_in[2];
  const float* Whh = (const float*)d_in[3];
  const float* bih = (const float*)d_in[4];
  const float* bhh = (const float*)d_in[5];
  const float* W1  = (const float*)d_in[6];
  const float* gW0 = (const float*)d_in[7];
  const float* gb0 = (const float*)d_in[8];
  const float* gW1 = (const float*)d_in[9];
  const float* gb1 = (const float*)d_in[10];
  const float* W2  = (const float*)d_in[11];
  const int* uids  = (const int*)d_in[12];
  const int* seqs  = (const int*)d_in[13];
  const int* lens  = (const int*)d_in[14];
  const int* sidx  = (const int*)d_in[15];
  const int* src0  = (const int*)d_in[16];
  const int* dst0  = (const int*)d_in[17];
  const int* idx0  = (const int*)d_in[18];
  const int* src1  = (const int*)d_in[19];
  const int* dst1  = (const int*)d_in[20];
  const int* idx1  = (const int*)d_in[21];

  const int D = DV;
  const int N0 = in_sizes[12];
  const int T  = in_sizes[13] / N0;
  const int N1 = in_sizes[18];
  const int N2 = in_sizes[15];
  const int E0 = in_sizes[16];
  const int E1 = in_sizes[19];
  const int NITEM = in_sizes[1] / D - 1;   // rows 1..NI of item_emb

  // workspace carve-out
  char* p = (char*)d_ws;
  auto alloc = [&](size_t bytes) -> char* {
    char* r = p; p += (bytes + 255) & ~(size_t)255; return r;
  };
  _Float16* Wcat  = (_Float16*)alloc((size_t)4 * D * 2 * D * 2);
  _Float16* W1h   = (_Float16*)alloc((size_t)D * 2 * D * 2);
  _Float16* gW0h  = (_Float16*)alloc((size_t)D * D * 2);
  _Float16* gW1h  = (_Float16*)alloc((size_t)D * D * 2);
  _Float16* W2h   = (_Float16*)alloc((size_t)D * 2 * D * 2);
  _Float16* it1n  = (_Float16*)alloc((size_t)NITEM * D * 2);
  _Float16* srh   = (_Float16*)alloc((size_t)N2 * D * 2);
  float* biasb = (float*)alloc((size_t)4 * D * 4);
  float* hn    = (float*)alloc((size_t)N0 * D * 4);
  float* feat  = (float*)alloc((size_t)N0 * D * 4);
  float* cur   = (float*)alloc((size_t)N2 * D * 4);
  float* feat1 = (float*)alloc((size_t)N1 * D * 4);
  float* feat2 = (float*)alloc((size_t)N2 * D * 4);
  float* score = (float*)alloc((size_t)E0 * 4);
  float* wbuf  = (float*)alloc((size_t)E0 * 4);
  float* mbuf  = (float*)alloc((size_t)N1 * 4);
  float* zbuf  = (float*)alloc((size_t)N1 * 4);
  float* agg   = (float*)alloc((size_t)N1 * D * 4);

  const float NEGINF = -__builtin_inff();

  // weight prep
  build_wcat<<<(4 * D * 2 * D + 255) / 256, 256, 0, stream>>>(Wih, Whh, Wcat);
  cvt_f32_to_f16<<<(D * 2 * D + 255) / 256, 256, 0, stream>>>(W1, W1h, D * 2 * D);
  cvt_f32_to_f16<<<(D * D + 255) / 256, 256, 0, stream>>>(gW0, gW0h, D * D);
  cvt_f32_to_f16<<<(D * D + 255) / 256, 256, 0, stream>>>(gW1, gW1h, D * D);
  cvt_f32_to_f16<<<(D * 2 * D + 255) / 256, 256, 0, stream>>>(W2, W2h, D * 2 * D);
  bias_sum<<<(4 * D + 255) / 256, 256, 0, stream>>>(bih, bhh, biasb, 4 * D);
  renorm_rows_f16<<<(NITEM + 15) / 16, 256, 0, stream>>>(item_emb + D, it1n, NITEM);

  // LSTM over sequences
  lstm_kernel<<<N0 / 16, 256, 0, stream>>>(item_emb, seqs, lens, Wcat, biasb, hn, T);

  // feat = relu([renorm(user), hn] @ W1^T); override rows at cur_sidx with hn
  feat_kernel<<<N0 / 16, 256, 0, stream>>>(user_emb, uids, hn, W1h, feat);
  cur_kernel<<<(N2 * D + 255) / 256, 256, 0, stream>>>(hn, sidx, cur, feat, N2);

  // GAT layer 0: feat(N0) -> feat1(N1)
  fill_f32<<<(N1 + 255) / 256, 256, 0, stream>>>(mbuf, NEGINF, N1);
  fill_f32<<<(N1 + 255) / 256, 256, 0, stream>>>(zbuf, 0.f, N1);
  fill_f32<<<(N1 * D + 255) / 256, 256, 0, stream>>>(agg, 0.f, N1 * D);
  edge_score_kernel<<<(E0 + 7) / 8, 256, 0, stream>>>(feat, src0, dst0, idx0, score, mbuf, E0);
  edge_w_kernel<<<(E0 + 255) / 256, 256, 0, stream>>>(score, dst0, mbuf, wbuf, zbuf, E0);
  edge_agg_kernel<<<(E0 + 7) / 8, 256, 0, stream>>>(feat, src0, dst0, wbuf, zbuf, agg, E0);
  gat_out_kernel<<<N1 / 16, 256, 0, stream>>>(agg, feat, idx0, gW0h, gb0, feat1);

  // GAT layer 1: feat1(N1) -> feat2(N2)
  fill_f32<<<(N2 + 255) / 256, 256, 0, stream>>>(mbuf, NEGINF, N2);
  fill_f32<<<(N2 + 255) / 256, 256, 0, stream>>>(zbuf, 0.f, N2);
  fill_f32<<<(N2 * D + 255) / 256, 256, 0, stream>>>(agg, 0.f, N2 * D);
  edge_score_kernel<<<(E1 + 7) / 8, 256, 0, stream>>>(feat1, src1, dst1, idx1, score, mbuf, E1);
  edge_w_kernel<<<(E1 + 255) / 256, 256, 0, stream>>>(score, dst1, mbuf, wbuf, zbuf, E1);
  edge_agg_kernel<<<(E1 + 7) / 8, 256, 0, stream>>>(feat1, src1, dst1, wbuf, zbuf, agg, E1);
  gat_out_kernel<<<N2 / 16, 256, 0, stream>>>(agg, feat1, idx1, gW1h, gb1, feat2);

  // sr = [cur, feat2] @ W2^T  (f16), then logits = sr @ it1n^T
  sr_kernel<<<N2 / 16, 256, 0, stream>>>(cur, feat2, W2h, srh);
  logits_kernel<<<dim3(N2 / 16, (NITEM + 127) / 128), 256, 0, stream>>>(
      srh, it1n, (float*)d_out, NITEM);
}
